// Head_18743237280301
// MI455X (gfx1250) — matrix operand order repaired and checked by reading, not compiled, not run on hardware
//
#include <hip/hip_runtime.h>

#if __has_include(<hip/amd_detail/amd_gfx1250_TDM.h>)
#define ATTN_TDM_6ARG 1
#else
#define ATTN_TDM_6ARG 0
#endif

#if __has_builtin(__builtin_amdgcn_tensor_load_to_lds) && \
    __has_builtin(__builtin_amdgcn_s_wait_tensorcnt)
#define ATTN_USE_TDM 1
#else
#define ATTN_USE_TDM 0
#endif

typedef __attribute__((ext_vector_type(16))) _Float16 v16h;
typedef __attribute__((ext_vector_type(8)))  _Float16 v8h;
typedef __attribute__((ext_vector_type(8)))  float    v8f;
typedef __attribute__((ext_vector_type(4)))  unsigned v4u;
typedef __attribute__((ext_vector_type(8)))  int      v8i;
typedef __attribute__((ext_vector_type(4)))  int      v4i;

#define ATTN_B  8
#define ATTN_T  2048
#define ATTN_D  1024
#define ATTN_HS 128

#define KBLK 64        // keys per staged block
#define KROW 136       // K tile LDS row stride: 128 + 8 pad halves (bank stagger)
#define VROW 72        // V tile LDS row stride: 64 + 8 pad halves
#define PROW 72        // P transpose LDS row stride: 64 + 8 pad halves

__device__ __forceinline__ v8f wmma_f16(v16h a, v16h b, v8f c) {
  return __builtin_amdgcn_wmma_f32_16x16x32_f16(false, a, false, b, (short)0, c,
                                                false, false);
}

__device__ __forceinline__ v16h cat8(v8h lo, v8h hi) {
  return __builtin_shufflevector(lo, hi, 0, 1, 2, 3, 4, 5, 6, 7, 8, 9, 10, 11,
                                 12, 13, 14, 15);
}

__device__ __forceinline__ v16h ldfrag(const _Float16* p, int gap) {
  return cat8(*(const v8h*)p, *(const v8h*)(p + gap));
}

#if ATTN_USE_TDM
// 2-D tile DMA: global -> LDS, f16 elements, optional LDS row padding.
// D# layout per CDNA5 ISA.  Tiles are constructed
// fully in-bounds, so tensor dims == tile dims (no OOB clamping needed).
__device__ __forceinline__ void tdm_load_2d(unsigned lds_byte_off,
                                            const void* gsrc, unsigned tile_d0,
                                            unsigned tile_d1, unsigned stride_d0,
                                            unsigned pad_int_code,
                                            unsigned pad_amt_code,
                                            unsigned pad_en) {
  unsigned long long ga = (unsigned long long)(size_t)gsrc;
  v4u g0 = {1u,                                   // count=1, user descriptor
            lds_byte_off,                         // lds_addr
            (unsigned)(ga & 0xffffffffull),       // global_addr[31:0]
            (unsigned)((ga >> 32) & 0x1ffffffull) // global_addr[56:32]
                | (2u << 30)};                    // type=2 ("image")
  v8i g1 = {(int)((1u << 16)                      // data_size = 2 bytes
                  | (pad_en << 20) | (pad_int_code << 22) | (pad_amt_code << 25)),
            (int)((tile_d0 & 0xffffu) << 16),                  // tensor_dim0 lo
            (int)(((tile_d0 >> 16) & 0xffffu) |
                  ((tile_d1 & 0xffffu) << 16)),                // dim0 hi | dim1 lo
            (int)(((tile_d1 >> 16) & 0xffffu) |
                  ((tile_d0 & 0xffffu) << 16)),                // dim1 hi | tile_dim0
            (int)(tile_d1 & 0xffffu),                          // tile_dim1 (dim2=0)
            (int)stride_d0,                                    // dim0 stride lo
            0, 0};
  v4i z4 = {0, 0, 0, 0};
#if ATTN_TDM_6ARG
  v8i z8 = {0, 0, 0, 0, 0, 0, 0, 0};
  __builtin_amdgcn_tensor_load_to_lds(g0, g1, z4, z4, z8, 0);
#else
  __builtin_amdgcn_tensor_load_to_lds(g0, g1, z4, z4, 0);
#endif
}
#endif  // ATTN_USE_TDM

// ---------------------------------------------------------------------------
// Pass 0: one-time weight prep.  W[D][HS] f32 -> wt[mat][HS][D] f16 so pass-1
// B-fragments are contiguous 16-byte loads per lane.
// ---------------------------------------------------------------------------
__global__ __launch_bounds__(256) void wprep_kernel(
    const float* __restrict__ Wq, const float* __restrict__ Wk,
    const float* __restrict__ Wv, _Float16* __restrict__ wt) {
  const int idx = blockIdx.x * 256 + threadIdx.x;  // [0, 3*HS*D)
  const int m = idx >> 17;                         // HS*D = 1<<17
  const int rem = idx & ((ATTN_HS * ATTN_D) - 1);
  const int n = rem >> 10;  // hs column
  const int d = rem & (ATTN_D - 1);
  const float* W = (m == 0) ? Wq : (m == 1) ? Wk : Wv;
  wt[idx] = (_Float16)W[d * ATTN_HS + n];
}

// ---------------------------------------------------------------------------
// Pass 1: fused QKV projection.  6 waves: {row strip 0/1} x {q,k,v}.
// Software-pipelined: next A chunk (f32) and next B fragment are issued
// before the current WMMA so s_wait_loadcnt overlaps compute.
// ---------------------------------------------------------------------------
__global__ __launch_bounds__(192) void qkv_proj_kernel(
    const float* __restrict__ x, const _Float16* __restrict__ wt,
    _Float16* __restrict__ qo, _Float16* __restrict__ ko,
    _Float16* __restrict__ vto) {
  const int wave = threadIdx.x >> 5;
  const int lane = threadIdx.x & 31;
  const int half = lane >> 4;
  const int lr = lane & 15;
  const int strip = wave & 1;
  const int mat = wave >> 1;  // 0=q, 1=k, 2=v
  const int row0 = blockIdx.x * 32 + strip * 16;
  const _Float16* wmat = wt + (size_t)mat * ATTN_HS * ATTN_D;

  v8f acc[8];
#pragma unroll
  for (int n = 0; n < 8; ++n)
#pragma unroll
    for (int e = 0; e < 8; ++e) acc[n][e] = 0.0f;

  const float* aprow = x + (size_t)(row0 + lr) * ATTN_D + half * 8;
  v8f f0 = *(const v8f*)(aprow);
  v8f f1 = *(const v8f*)(aprow + 16);

  for (int kb = 0; kb < ATTN_D; kb += 32) {
    v16h a;
#pragma unroll
    for (int e = 0; e < 8; ++e) {
      a[e] = (_Float16)f0[e];
      a[8 + e] = (_Float16)f1[e];
    }
    if (kb + 32 < ATTN_D) {  // issue next A chunk early; prefetch one ahead
      f0 = *(const v8f*)(aprow + kb + 32);
      f1 = *(const v8f*)(aprow + kb + 48);
      __builtin_prefetch(aprow + kb + 64, 0, 3);
    }
    const _Float16* wrow = wmat + kb + half * 8;
    v16h bf = ldfrag(wrow + (size_t)lr * ATTN_D, 16);
#pragma unroll
    for (int n = 0; n < 8; ++n) {
      v16h bnext = bf;
      if (n < 7)
        bnext = ldfrag(wrow + (size_t)((n + 1) * 16 + lr) * ATTN_D, 16);
      acc[n] = wmma_f16(a, bf, acc[n]);
      bf = bnext;
    }
  }

  if (mat == 2) {  // v stored transposed per batch: vT[b][hs][t]
    const int bidx = row0 / ATTN_T;
    const int r = row0 % ATTN_T;
    _Float16* dst = vto + (size_t)bidx * ATTN_HS * ATTN_T;
#pragma unroll
    for (int n = 0; n < 8; ++n)
#pragma unroll
      for (int i = 0; i < 8; ++i)
        dst[(size_t)(n * 16 + lr) * ATTN_T + r + half * 8 + i] =
            (_Float16)acc[n][i];
  } else {
    _Float16* dst = (mat == 0 ? qo : ko) + (size_t)row0 * ATTN_HS;
#pragma unroll
    for (int n = 0; n < 8; ++n)
#pragma unroll
      for (int i = 0; i < 8; ++i)
        dst[(size_t)(half * 8 + i) * ATTN_HS + n * 16 + lr] =
            (_Float16)acc[n][i];
  }
}

// ---------------------------------------------------------------------------
// Pass 2: causal flash attention.  4 waves x 16 query rows per block share
// 64-key K/V tiles staged in LDS, double-buffered by the Tensor Data Mover.
// Blocks are scheduled longest-first (reverse tile order) to shrink the tail.
// ---------------------------------------------------------------------------
__global__ __launch_bounds__(128) void flash_attn_kernel(
    const _Float16* __restrict__ q, const _Float16* __restrict__ k,
    const _Float16* __restrict__ vt, float* __restrict__ out) {
  __shared__ __align__(16) _Float16 ktile[2 * KBLK * KROW];     // 34 KB
  __shared__ __align__(16) _Float16 vtile[2 * ATTN_HS * VROW];  // 36 KB
  __shared__ __align__(16) _Float16 pshare[4 * 16 * PROW];      // 9 KB

  const int tid = threadIdx.x;
  const int wave = tid >> 5;
  const int lane = tid & 31;
  const int half = lane >> 4;
  const int lr = lane & 15;
  const int b = blockIdx.x;
  const int qy = (int)gridDim.y - 1 - (int)blockIdx.y;  // longest first
  const int rblk = qy * 64;
  const int r0 = rblk + wave * 16;
  const int nkb = qy + 1;  // 64-key blocks covering keys < rblk+64

  const _Float16* qb = q + (size_t)b * ATTN_T * ATTN_HS;
  const _Float16* kbp = k + (size_t)b * ATTN_T * ATTN_HS;
  const _Float16* vb = vt + (size_t)b * ATTN_HS * ATTN_T;
  _Float16* const pw = &pshare[wave * 16 * PROW];

  // Q A-fragments (4 HS chunks of 32), resident in VGPRs.
  v16h qa[4];
#pragma unroll
  for (int c = 0; c < 4; ++c)
    qa[c] = ldfrag(qb + (size_t)(r0 + lr) * ATTN_HS + c * 32 + half * 16, 8);

  v8f o[8];
#pragma unroll
  for (int n = 0; n < 8; ++n)
#pragma unroll
    for (int e = 0; e < 8; ++e) o[n][e] = 0.0f;
  float mrun[8], lrun[8];
#pragma unroll
  for (int i = 0; i < 8; ++i) {
    mrun[i] = -3.0e38f;
    lrun[i] = 0.0f;
  }
  const float scale = 0.08838834764831845f;  // 1/sqrt(128)

#if ATTN_USE_TDM
  if (wave == 0) {  // prologue: stage buffer 0
    tdm_load_2d((unsigned)(size_t)&ktile[0], kbp, ATTN_HS, KBLK, ATTN_HS,
                /*64 dwords*/ 5, /*4 dwords*/ 3, 1);
    tdm_load_2d((unsigned)(size_t)&vtile[0], vb, KBLK, ATTN_HS, ATTN_T,
                /*32 dwords*/ 4, /*4 dwords*/ 3, 1);
  }
#endif

  for (int jb = 0; jb < nkb; ++jb) {
    const int j0 = jb * KBLK;
    const int cur = jb & 1;
    const _Float16* kt = &ktile[cur * KBLK * KROW];
    const _Float16* vl = &vtile[cur * ATTN_HS * VROW];

#if ATTN_USE_TDM
    if (wave == 0) {
      if (jb + 1 < nkb) {  // prefetch next buffer, then wait for current
        const int nxt = cur ^ 1;
        tdm_load_2d((unsigned)(size_t)&ktile[nxt * KBLK * KROW],
                    kbp + (size_t)(j0 + KBLK) * ATTN_HS, ATTN_HS, KBLK, ATTN_HS,
                    5, 3, 1);
        tdm_load_2d((unsigned)(size_t)&vtile[nxt * ATTN_HS * VROW],
                    vb + j0 + KBLK, KBLK, ATTN_HS, ATTN_T, 4, 3, 1);
        __builtin_amdgcn_s_wait_tensorcnt(2);
      } else {
        __builtin_amdgcn_s_wait_tensorcnt(0);
      }
    }
    __syncthreads();
#else
    // fallback: synchronous cooperative staging with vector loads
    for (int u = tid; u < (KBLK * ATTN_HS) / 8; u += 128) {
      const int r = u >> 4, c = u & 15;
      *(v8h*)&ktile[cur * KBLK * KROW + r * KROW + c * 8] =
          *(const v8h*)(kbp + (size_t)(j0 + r) * ATTN_HS + c * 8);
    }
    for (int u = tid; u < (ATTN_HS * KBLK) / 8; u += 128) {
      const int r = u >> 3, c = u & 7;
      *(v8h*)&vtile[cur * ATTN_HS * VROW + r * VROW + c * 8] =
          *(const v8h*)(vb + (size_t)r * ATTN_T + j0 + c * 8);
    }
    __syncthreads();
#endif

    // ---- S = Q·K^T : 16 fragments, rotated so ds_load overlaps wmma ----
    v8f s[4];
#pragma unroll
    for (int t = 0; t < 4; ++t)
#pragma unroll
      for (int e = 0; e < 8; ++e) s[t][e] = 0.0f;
    {
      v16h bf = ldfrag(kt + lr * KROW + half * 16, 8);  // c=0, t=0
#pragma unroll
      for (int u = 0; u < 16; ++u) {
        const int c = u >> 2, t = u & 3;
        v16h bnext = bf;
        if (u < 15) {
          const int c2 = (u + 1) >> 2, t2 = (u + 1) & 3;
          bnext = ldfrag(kt + (t2 * 16 + lr) * KROW + c2 * 32 + half * 16, 8);
        }
        s[t] = wmma_f16(qa[c], bf, s[t]);
        bf = bnext;
      }
    }

    // ---- causal mask + online softmax ----
#pragma unroll
    for (int i = 0; i < 8; ++i) {
      const int qrow = r0 + half * 8 + i;
      float vv[4];
#pragma unroll
      for (int t = 0; t < 4; ++t) {
        vv[t] = s[t][i] * scale;
        if (j0 + t * 16 + lr > qrow) vv[t] = -3.0e38f;
      }
      float mx = fmaxf(fmaxf(vv[0], vv[1]), fmaxf(vv[2], vv[3]));
#pragma unroll
      for (int off = 1; off < 16; off <<= 1)
        mx = fmaxf(mx, __shfl_xor(mx, off, 16));
      const float mnew = fmaxf(mrun[i], mx);
      const float corr = __expf(mrun[i] - mnew);
      mrun[i] = mnew;
      float rsum = 0.0f;
#pragma unroll
      for (int t = 0; t < 4; ++t) {
        const float p = __expf(vv[t] - mnew);
        s[t][i] = p;
        rsum += p;
      }
#pragma unroll
      for (int off = 1; off < 16; off <<= 1) rsum += __shfl_xor(rsum, off, 16);
      lrun[i] = lrun[i] * corr + rsum;
#pragma unroll
      for (int n = 0; n < 8; ++n) o[n][i] *= corr;
    }

    // ---- P: C/D layout -> A layout via wave-private LDS ----
#pragma unroll
    for (int t = 0; t < 4; ++t)
#pragma unroll
      for (int i = 0; i < 8; ++i)
        pw[(half * 8 + i) * PROW + t * 16 + lr] = (_Float16)s[t][i];
    __syncthreads();
    v16h pa[2];
#pragma unroll
    for (int a = 0; a < 2; ++a)
      pa[a] = ldfrag(pw + lr * PROW + a * 32 + half * 16, 8);

    // ---- O += P·V : 16 fragments, rotated ----
    {
      v16h vf = ldfrag(vl + lr * VROW + half * 16, 8);  // n=0, a=0
#pragma unroll
      for (int u = 0; u < 16; ++u) {
        const int n = u >> 1, a = u & 1;
        v16h vnext = vf;
        if (u < 15) {
          const int n2 = (u + 1) >> 1, a2 = (u + 1) & 1;
          bnext_label:
          vnext = ldfrag(vl + (n2 * 16 + lr) * VROW + a2 * 32 + half * 16, 8);
        }
        o[n] = wmma_f16(pa[a], vf, o[n]);
        vf = vnext;
      }
    }
    __syncthreads();  // all waves done with this buffer before it is reused
  }

  // ---- epilogue: normalize and store f32 ----
#pragma unroll
  for (int i = 0; i < 8; ++i) {
    const float inv = 1.0f / lrun[i];
    float* dst = out + ((size_t)b * ATTN_T + r0 + half * 8 + i) * ATTN_HS;
#pragma unroll
    for (int n = 0; n < 8; ++n) dst[n * 16 + lr] = o[n][i] * inv;
  }
}

extern "C" void kernel_launch(void* const* d_in, const int* in_sizes, int n_in,
                              void* d_out, int out_size, void* d_ws,
                              size_t ws_size, hipStream_t stream) {
  (void)in_sizes;
  (void)n_in;
  (void)out_size;
  (void)ws_size;
  const float* x = (const float*)d_in[0];
  const float* Wq = (const float*)d_in[1];
  const float* Wk = (const float*)d_in[2];
  const float* Wv = (const float*)d_in[3];
  float* out = (float*)d_out;

  const size_t elems = (size_t)ATTN_B * ATTN_T * ATTN_HS;  // 2M f16 = 4MB each
  _Float16* qbuf = (_Float16*)d_ws;
  _Float16* kbuf = qbuf + elems;
  _Float16* vtbuf = kbuf + elems;
  _Float16* wtbuf = vtbuf + elems;  // 3*128*1024 f16 = 768 KB

  wprep_kernel<<<(3 * ATTN_HS * ATTN_D) / 256, 256, 0, stream>>>(Wq, Wk, Wv,
                                                                 wtbuf);
  dim3 g1((ATTN_B * ATTN_T) / 32);  // 512 blocks x 6 waves
  qkv_proj_kernel<<<g1, 192, 0, stream>>>(x, wtbuf, qbuf, kbuf, vtbuf);
  dim3 g2(ATTN_B, ATTN_T / 64);  // 256 blocks x 4 waves, longest-first order
  flash_attn_kernel<<<g2, 128, 0, stream>>>(qbuf, kbuf, vtbuf, out);
}
